// Bigram_30863634989142
// MI455X (gfx1250) — matrix operand order repaired and checked by reading, not compiled, not run on hardware
//
#include <hip/hip_runtime.h>
#include <stdint.h>

// ---------------------------------------------------------------------------
// Model dims (compile-time constants from the reference)
// ---------------------------------------------------------------------------
#define VOCAB  96
#define VPAD   128      // VOCAB padded to N-tile multiple
#define EMBED  192
#define HIDDEN 768      // 4*EMBED
#define BATCH  256
#define SEQ    128
#define NTOK   (BATCH * SEQ)     // 32768 rows
#define LAYERS 6
#define EPS    1e-5f

typedef _Float16 v16h __attribute__((ext_vector_type(16)));
typedef _Float16 v8h  __attribute__((ext_vector_type(8)));
typedef float    v8f  __attribute__((ext_vector_type(8)));

// LDS row strides (in halves). 48 halves = 96 bytes = multiple of 32B, so all
// v8h/v16h fragment loads below are naturally aligned.
#define LDA 48
#define LDB 48

// ---------------------------------------------------------------------------
// CDNA5 async global->LDS copies (ASYNCcnt-tracked; CDNA5 ISA).
// The instruction offset applies to BOTH the LDS dest and the global source,
// so one (ldsOff, gptr) pair can drive multiple b128 copies.
// ---------------------------------------------------------------------------
__device__ __forceinline__ void async_copy_16B(uint32_t ldsOff, const void* gptr) {
  asm volatile("global_load_async_to_lds_b128 %0, %1, off"
               :: "v"(ldsOff), "v"(gptr) : "memory");
}
__device__ __forceinline__ void async_copy_32B(uint32_t ldsOff, const void* gptr) {
  asm volatile("global_load_async_to_lds_b128 %0, %1, off\n\t"
               "global_load_async_to_lds_b128 %0, %1, off offset:16"
               :: "v"(ldsOff), "v"(gptr) : "memory");
}
__device__ __forceinline__ void wait_async_all() {
  asm volatile("s_wait_asynccnt 0x0" ::: "memory");
}

// ---------------------------------------------------------------------------
// Weight prep: f32 [K,N] -> f16 transposed [Npad,K], zero-padding rows n>=N.
// Lets the GEMM B-stage be pure row-major async copies with no guards.
// ---------------------------------------------------------------------------
__global__ __launch_bounds__(256) void convt_kernel(const float* __restrict__ in,
                                                    _Float16* __restrict__ out,
                                                    int K, int N, int Npad) {
  int i = blockIdx.x * 256 + threadIdx.x;        // over Npad*K
  if (i >= Npad * K) return;
  int n = i / K;
  int k = i - n * K;
  out[i] = (n < N) ? (_Float16)in[(size_t)k * N + n] : (_Float16)0.f;
}

// ---------------------------------------------------------------------------
// Embedding: x[row, e] = tok_emb[index[row], e] + pos_emb[row % SEQ, e]
// ---------------------------------------------------------------------------
__global__ __launch_bounds__(256) void embed_kernel(const int* __restrict__ index,
                                                    const float* __restrict__ tok,
                                                    const float* __restrict__ pos,
                                                    float* __restrict__ x) {
  int i = blockIdx.x * 256 + threadIdx.x;        // over NTOK*EMBED
  if (i >= NTOK * EMBED) return;
  int row = i / EMBED;
  int e   = i - row * EMBED;
  int t   = row & (SEQ - 1);
  x[i] = tok[(size_t)index[row] * EMBED + e] + pos[(size_t)t * EMBED + e];
}

// ---------------------------------------------------------------------------
// LayerNorm (optionally applied twice: n2 = LN(LN(x))) -> f16 output.
// One wave32 per row; EMBED=192 -> 6 elements per lane; shfl_xor reductions.
// ---------------------------------------------------------------------------
__global__ __launch_bounds__(256) void ln_kernel(const float* __restrict__ x,
                                                 const float* __restrict__ g1,
                                                 const float* __restrict__ b1,
                                                 const float* __restrict__ g2,
                                                 const float* __restrict__ b2,
                                                 _Float16* __restrict__ out,
                                                 int rows, int doDouble) {
  int row  = blockIdx.x * 8 + (threadIdx.x >> 5);
  int lane = threadIdx.x & 31;
  if (row >= rows) return;
  const float* xr = x + (size_t)row * EMBED;

  float v[6];
  float s = 0.f;
#pragma unroll
  for (int j = 0; j < 6; ++j) { v[j] = xr[lane + j * 32]; s += v[j]; }
#pragma unroll
  for (int off = 16; off; off >>= 1) s += __shfl_xor(s, off, 32);
  float mean = s * (1.f / EMBED);
  float vs = 0.f;
#pragma unroll
  for (int j = 0; j < 6; ++j) { float d = v[j] - mean; vs += d * d; }
#pragma unroll
  for (int off = 16; off; off >>= 1) vs += __shfl_xor(vs, off, 32);
  float rs = rsqrtf(vs * (1.f / EMBED) + EPS);
#pragma unroll
  for (int j = 0; j < 6; ++j) {
    int e = lane + j * 32;
    v[j] = (v[j] - mean) * rs * g1[e] + b1[e];
  }
  if (doDouble) {
    s = 0.f;
#pragma unroll
    for (int j = 0; j < 6; ++j) s += v[j];
#pragma unroll
    for (int off = 16; off; off >>= 1) s += __shfl_xor(s, off, 32);
    mean = s * (1.f / EMBED);
    vs = 0.f;
#pragma unroll
    for (int j = 0; j < 6; ++j) { float d = v[j] - mean; vs += d * d; }
#pragma unroll
    for (int off = 16; off; off >>= 1) vs += __shfl_xor(vs, off, 32);
    rs = rsqrtf(vs * (1.f / EMBED) + EPS);
#pragma unroll
    for (int j = 0; j < 6; ++j) {
      int e = lane + j * 32;
      v[j] = (v[j] - mean) * rs * g2[e] + b2[e];
    }
  }
  _Float16* orow = out + (size_t)row * EMBED;
#pragma unroll
  for (int j = 0; j < 6; ++j) orow[lane + j * 32] = (_Float16)v[j];
}

// ---------------------------------------------------------------------------
// WMMA GEMM: C[M,N] = act(A[M,K] * Bt[N,K]^T + bias[N])
//   A:  f16 [M,K] row-major.
//   Bt: f16 [Npad,K] row-major (weights pre-transposed & N-padded, so B staging
//       is unconditional row-major async copy; Npad >= gridDim.x*64).
//   C:  f16 (OUT_F16) or f32. act = ReLU if RELU.
//   Block tile 128x64, 8 waves (4x2), each wave 32x32 = 2x2 wmma fragments.
//   K, N compile-time; M multiple of 128 at runtime.
//   Double-buffered LDS + async copies: copy for step k+1 overlaps the 4 WMMAs
//   of step k; one barrier per K-step.
// Fragment layouts (wave32; A/B use this kernel's contiguous k order, C/D as in ISA 7.12.2):
//   A 16x32 f16: lane m=lane&15, halves 0..15 -> K = (lane>>4)*16 + 0..15
//   B 32x16 f16: lane n=lane&15, halves 0..15 -> K = (lane>>4)*16 + 0..15
//   C/D f32: VGPR r -> M = r + (lane>>4)*8, N = lane&15
// ---------------------------------------------------------------------------
template <int RELU, int OUT_F16, int K, int N>
__global__ __launch_bounds__(256) void gemm_wmma_kernel(const _Float16* __restrict__ A,
                                                        const _Float16* __restrict__ Bt,
                                                        const float* __restrict__ bias,
                                                        void* __restrict__ Cout, int M) {
  __shared__ _Float16 ldsA[2][128 * LDA];   // A tiles, row-major [m][k], padded
  __shared__ _Float16 ldsB[2][64 * LDB];    // B tiles, row-major [n][k], padded

  const int tid     = threadIdx.x;
  const int lane    = tid & 31;
  const int wave    = tid >> 5;
  const int waveM   = wave >> 1;         // 0..3
  const int waveN   = wave & 1;          // 0..1
  const int mBase   = blockIdx.y * 128;
  const int nBase   = blockIdx.x * 64;
  const int halfSel = lane >> 4;         // 0/1
  const int l16     = lane & 15;

  // Per-thread staging chunks (async global->LDS):
  //   A tile 128x32 halves: thread -> row ar, 32B chunk at ac.
  //   B tile  64x32 halves: thread -> row bn, 16B chunk at bc.
  const int ar = tid >> 1;
  const int ac = (tid & 1) * 16;
  const int bn = tid >> 2;
  const int bc = (tid & 3) * 8;
  const uint32_t aLds[2] = { (uint32_t)(uintptr_t)&ldsA[0][ar * LDA + ac],
                             (uint32_t)(uintptr_t)&ldsA[1][ar * LDA + ac] };
  const uint32_t bLds[2] = { (uint32_t)(uintptr_t)&ldsB[0][bn * LDB + bc],
                             (uint32_t)(uintptr_t)&ldsB[1][bn * LDB + bc] };
  const _Float16* aSrc = A  + (size_t)(mBase + ar) * K + ac;
  const _Float16* bSrc = Bt + (size_t)(nBase + bn) * K + bc;

  constexpr int STEPS = K / 32;
  v8f acc[2][2] = {};

  // prologue: stage step 0 into buffer 0
  async_copy_32B(aLds[0], aSrc);
  async_copy_16B(bLds[0], bSrc);

#pragma unroll
  for (int kt = 0; kt < STEPS; ++kt) {
    const int cur = kt & 1;
    wait_async_all();                  // copy into buf[cur] complete
    __syncthreads();                   // ... and visible to all waves; prior
                                       // reads of buf[cur^1] are also done.
    if (kt + 1 < STEPS) {              // overlap next copy with this compute
      async_copy_32B(aLds[cur ^ 1], aSrc + (kt + 1) * 32);
      async_copy_16B(bLds[cur ^ 1], bSrc + (kt + 1) * 32);
    }

    const _Float16* lA = ldsA[cur];
    const _Float16* lB = ldsB[cur];
    const int mW = waveM * 32;
    const int nW = waveN * 32;
#pragma unroll
    for (int fm = 0; fm < 2; ++fm) {
      const int mrow = mW + fm * 16 + l16;
      v8h lo = *(const v8h*)(&lA[mrow * LDA + halfSel * 16]);
      v8h hi = *(const v8h*)(&lA[mrow * LDA + halfSel * 16 + 8]);
      v16h a;
#pragma unroll
      for (int i = 0; i < 8; ++i) { a[i] = lo[i]; a[i + 8] = hi[i]; }
#pragma unroll
      for (int fn = 0; fn < 2; ++fn) {
        const int ncol = nW + fn * 16 + l16;
        v16h b = *(const v16h*)(&lB[ncol * LDB + halfSel * 16]);
        acc[fm][fn] = __builtin_amdgcn_wmma_f32_16x16x32_f16(
            false, a, false, b, (short)0, acc[fm][fn], false, false);
      }
    }
  }

  // --- epilogue: bias (+ReLU), store f16 or f32 (compile-time N offsets) ---
  float bv[2];
  int   nIdx[2];
#pragma unroll
  for (int fn = 0; fn < 2; ++fn) {
    nIdx[fn] = nBase + waveN * 32 + fn * 16 + l16;
    bv[fn]   = (nIdx[fn] < N) ? bias[nIdx[fn]] : 0.f;
  }
#pragma unroll
  for (int fm = 0; fm < 2; ++fm) {
#pragma unroll
    for (int fn = 0; fn < 2; ++fn) {
      if (nIdx[fn] >= N) continue;
      const int mRow0 = mBase + waveM * 32 + fm * 16 + halfSel * 8;
#pragma unroll
      for (int r = 0; r < 8; ++r) {
        float v = acc[fm][fn][r] + bv[fn];
        if (RELU) v = v > 0.f ? v : 0.f;
        size_t idx = (size_t)(mRow0 + r) * N + nIdx[fn];
        if (OUT_F16) ((_Float16*)Cout)[idx] = (_Float16)v;
        else         ((float*)Cout)[idx] = v;
      }
    }
  }
}

// ---------------------------------------------------------------------------
// Per-row NLL: rowloss[row] = logsumexp(pred[row,:]) - pred[row, tgt[row]]
// One wave32 per row; VOCAB=96 -> 3 elements per lane.
// ---------------------------------------------------------------------------
__global__ __launch_bounds__(256) void rowloss_kernel(const float* __restrict__ pred,
                                                      const int* __restrict__ tgt,
                                                      float* __restrict__ rowloss) {
  int row  = blockIdx.x * 8 + (threadIdx.x >> 5);
  int lane = threadIdx.x & 31;
  if (row >= NTOK) return;
  const float* p = pred + (size_t)row * VOCAB;
  float v[3];
  float mx = -3.0e38f;
#pragma unroll
  for (int j = 0; j < 3; ++j) { v[j] = p[lane + j * 32]; mx = fmaxf(mx, v[j]); }
#pragma unroll
  for (int off = 16; off; off >>= 1) mx = fmaxf(mx, __shfl_xor(mx, off, 32));
  float se = 0.f;
#pragma unroll
  for (int j = 0; j < 3; ++j) se += __expf(v[j] - mx);
#pragma unroll
  for (int off = 16; off; off >>= 1) se += __shfl_xor(se, off, 32);
  int t = tgt[row];
  float vt = 0.f;
#pragma unroll
  for (int j = 0; j < 3; ++j) vt += (lane + j * 32 == t) ? v[j] : 0.f;
#pragma unroll
  for (int off = 16; off; off >>= 1) vt += __shfl_xor(vt, off, 32);
  if (lane == 0) rowloss[row] = __logf(se) + mx - vt;
}

// Single-block final reduction -> mean loss
__global__ __launch_bounds__(256) void reduce_kernel(const float* __restrict__ rowloss,
                                                     float* __restrict__ out, int n) {
  __shared__ float sm[256];
  float s = 0.f;
  for (int i = threadIdx.x; i < n; i += 256) s += rowloss[i];
  sm[threadIdx.x] = s;
  __syncthreads();
  for (int st = 128; st; st >>= 1) {
    if ((int)threadIdx.x < st) sm[threadIdx.x] += sm[threadIdx.x + st];
    __syncthreads();
  }
  if (threadIdx.x == 0) *out = sm[0] / (float)n;
}

// ---------------------------------------------------------------------------
// Launch
// ---------------------------------------------------------------------------
extern "C" void kernel_launch(void* const* d_in, const int* in_sizes, int n_in,
                              void* d_out, int out_size, void* d_ws, size_t ws_size,
                              hipStream_t stream) {
  (void)in_sizes; (void)n_in; (void)out_size; (void)ws_size;

  const int*   index   = (const int*)d_in[0];
  const int*   targets = (const int*)d_in[1];
  const float* tok_emb = (const float*)d_in[2];
  const float* pos_emb = (const float*)d_in[3];
  const float* ln1_g   = (const float*)d_in[4];
  const float* ln1_b   = (const float*)d_in[5];
  const float* ln2_g   = (const float*)d_in[6];
  const float* ln2_b   = (const float*)d_in[7];
  // d_in[8..10] = Wq, Wk, Wv -> dead code in the reference forward; skipped.
  const float* W1      = (const float*)d_in[11];
  const float* b1      = (const float*)d_in[12];
  const float* W2      = (const float*)d_in[13];
  const float* b2      = (const float*)d_in[14];
  const float* fn_g    = (const float*)d_in[15];
  const float* fn_b    = (const float*)d_in[16];
  const float* Wf      = (const float*)d_in[17];
  const float* bf      = (const float*)d_in[18];

  // Workspace carve-up (all offsets 256B-aligned)
  char* ws = (char*)d_ws;
  size_t off = 0;
  float*    X    = (float*)(ws + off);    off += (size_t)NTOK * EMBED * 4;     // 25.2 MB
  _Float16* A16  = (_Float16*)(ws + off); off += (size_t)NTOK * EMBED * 2;     // 12.6 MB
  _Float16* H16  = (_Float16*)(ws + off); off += (size_t)NTOK * HIDDEN * 2;    // 50.3 MB
  _Float16* W1t  = (_Float16*)(ws + off); off += (size_t)LAYERS * HIDDEN * EMBED * 2; // [768][192]/layer
  _Float16* W2t  = (_Float16*)(ws + off); off += (size_t)LAYERS * EMBED * HIDDEN * 2; // [192][768]/layer
  _Float16* Wft  = (_Float16*)(ws + off); off += (size_t)VPAD * EMBED * 2;            // [128][192]
  float*    rls  = (float*)(ws + off);    off += (size_t)NTOK * 4;

  float* pred = (float*)d_out;                   // [NTOK, VOCAB]
  float* loss = pred + (size_t)NTOK * VOCAB;     // single element after pred

  // 1) Convert + transpose (+pad) weights to f16 [Npad][K]
  for (int l = 0; l < LAYERS; ++l) {
    int n1 = HIDDEN * EMBED;   // W1[l]: [192,768] -> [768][192]
    convt_kernel<<<(n1 + 255) / 256, 256, 0, stream>>>(
        W1 + (size_t)l * EMBED * HIDDEN, W1t + (size_t)l * HIDDEN * EMBED,
        EMBED, HIDDEN, HIDDEN);
    int n2 = EMBED * HIDDEN;   // W2[l]: [768,192] -> [192][768]
    convt_kernel<<<(n2 + 255) / 256, 256, 0, stream>>>(
        W2 + (size_t)l * HIDDEN * EMBED, W2t + (size_t)l * EMBED * HIDDEN,
        HIDDEN, EMBED, EMBED);
  }
  {
    int n3 = VPAD * EMBED;     // Wf: [192,96] -> [128][192] (zero-padded rows)
    convt_kernel<<<(n3 + 255) / 256, 256, 0, stream>>>(Wf, Wft, EMBED, VOCAB, VPAD);
  }

  // 2) Embedding
  embed_kernel<<<(NTOK * EMBED + 255) / 256, 256, 0, stream>>>(index, tok_emb, pos_emb, X);

  // 3) Transformer layers (attention is dead code; block = FFN(LN(LN(x))))
  for (int l = 0; l < LAYERS; ++l) {
    ln_kernel<<<NTOK / 8, 256, 0, stream>>>(X, ln1_g + l * EMBED, ln1_b + l * EMBED,
                                            ln2_g + l * EMBED, ln2_b + l * EMBED,
                                            A16, NTOK, 1);
    // H = relu(A16 * W1 + b1)   [32768 x 768], f16 out
    gemm_wmma_kernel<1, 1, EMBED, HIDDEN><<<dim3(HIDDEN / 64, NTOK / 128), 256, 0, stream>>>(
        A16, W1t + (size_t)l * HIDDEN * EMBED, b1 + (size_t)l * HIDDEN,
        (void*)H16, NTOK);
    // X = H * W2 + b2           [32768 x 192], f32 out
    gemm_wmma_kernel<0, 0, HIDDEN, EMBED><<<dim3(EMBED / 64, NTOK / 128), 256, 0, stream>>>(
        H16, W2t + (size_t)l * EMBED * HIDDEN, b2 + (size_t)l * EMBED,
        (void*)X, NTOK);
  }

  // 4) Final LN -> f16
  ln_kernel<<<NTOK / 8, 256, 0, stream>>>(X, fn_g, fn_b, nullptr, nullptr, A16, NTOK, 0);

  // 5) Logits = A16 * Wf + bf  -> pred (f32, straight into d_out). N=96, Npad=128.
  gemm_wmma_kernel<0, 0, EMBED, VOCAB><<<dim3(VPAD / 64, NTOK / 128), 256, 0, stream>>>(
      A16, Wft, bf, (void*)pred, NTOK);

  // 6) Loss
  rowloss_kernel<<<NTOK / 8, 256, 0, stream>>>(pred, targets, rls);
  reduce_kernel<<<1, 256, 0, stream>>>(rls, loss, NTOK);
}
